// Attention_6743098655012
// MI455X (gfx1250) — compile-verified, matrix operand order checked by reading, not run on hardware
//
#include <hip/hip_runtime.h>

// ---------------------------------------------------------------------------
// MI455X (gfx1250) fused video-attention block, bf16 WMMA + fp32 accumulate.
// ---------------------------------------------------------------------------

typedef __attribute__((ext_vector_type(16))) __bf16 v16bf;
typedef __attribute__((ext_vector_type(8)))  float  v8f;

union V16 {
  v16bf v;
  __bf16 b[16];
  uint4 q[2];
};

union V8C {
  __bf16 b[8];
  uint4 q;
};

#define NTOK   1576
#define HEADS  12
#define HD     64
#define CDIM   768
#define BATCH  4
#define MROWS  (BATCH * NTOK)      // 6304
#define MT32   (MROWS / 32)        // 197 (exact)
#define AREA1  197
#define VSTR   1600                // padded token stride for V^T
#define QTILES 99                  // ceil(1576/16)
#define KITERS 50                  // ceil(1576/32)

static __device__ __forceinline__ __bf16 f2bf(float f) { return (__bf16)f; }

static __device__ __forceinline__ v8f zero8() {
  v8f z = {0.f, 0.f, 0.f, 0.f, 0.f, 0.f, 0.f, 0.f};
  return z;
}

static __device__ __forceinline__ v8f wmma_bf16(v16bf a, v16bf b, v8f c) {
  // emits v_wmma_f32_16x16x32_bf16
  return __builtin_amdgcn_wmma_f32_16x16x32_bf16(false, a, false, b, (short)0, c,
                                                 false, false);
}

// A/B operand slice for this lane from K-contiguous bf16 memory.
// ISA 7.12.2 16-bit 16x32 layout: lane half hl owns K = hl*8..hl*8+7 and
// 16+hl*8..16+hl*8+7  -> two contiguous 16B chunks.
static __device__ __forceinline__ v16bf load_bf16_tile(const __bf16* p, int hl) {
  V16 o;
  o.q[0] = *(const uint4*)(p + hl * 8);
  o.q[1] = *(const uint4*)(p + 16 + hl * 8);
  return o.v;
}

// ---------------------------------------------------------------------------
// Kernel C: one-shot fp32 -> bf16 conversion, 8 elements per thread.
// ---------------------------------------------------------------------------
__global__ __launch_bounds__(256) void st_cvt_kernel(const float* __restrict__ src,
                                                     __bf16* __restrict__ dst,
                                                     int n8) {
  int i = blockIdx.x * blockDim.x + threadIdx.x;
  if (i >= n8) return;
  const float4* s = (const float4*)src + (size_t)i * 2;
  float4 f0 = s[0], f1 = s[1];
  V8C o;
  o.b[0] = f2bf(f0.x); o.b[1] = f2bf(f0.y); o.b[2] = f2bf(f0.z); o.b[3] = f2bf(f0.w);
  o.b[4] = f2bf(f1.x); o.b[5] = f2bf(f1.y); o.b[6] = f2bf(f1.z); o.b[7] = f2bf(f1.w);
  ((uint4*)dst)[i] = o.q;
}

// ---------------------------------------------------------------------------
// Kernel 0: gathered spatial relative-position bias table [12][197][197] fp32
// ---------------------------------------------------------------------------
__global__ __launch_bounds__(256) void st_bias_kernel(const float* __restrict__ table,
                                                      float* __restrict__ bsp) {
  int t = blockIdx.x * blockDim.x + threadIdx.x;
  if (t >= AREA1 * AREA1) return;
  int a = t / AREA1, b2 = t % AREA1;
  int idx;
  if (a == 0 && b2 == 0)      idx = 731;           // nrd-1
  else if (a == 0)            idx = 729;           // nrd-3
  else if (b2 == 0)           idx = 730;           // nrd-2
  else {
    int s = a - 1, s2 = b2 - 1;
    int i1 = s / 14,  j1 = s % 14;
    int i2 = s2 / 14, j2 = s2 % 14;
    idx = (i1 - i2 + 13) * 27 + (j1 - j2 + 13);
  }
#pragma unroll
  for (int h = 0; h < HEADS; ++h)
    bsp[((size_t)h * AREA1 + a) * AREA1 + b2] = table[idx * HEADS + h];
}

// ---------------------------------------------------------------------------
// Kernel 1: QKV projection GEMM (NT, all-bf16 operands), per-wave 32(M)x64(N).
// Writes bf16 Q,K [B,H,N,64] and V^T [B,H,64,VSTR].
// ---------------------------------------------------------------------------
__global__ __launch_bounds__(256) void st_qkv_kernel(
    const __bf16* __restrict__ xbf, const __bf16* __restrict__ wbf,
    const float* __restrict__ qb, const float* __restrict__ vb,
    __bf16* __restrict__ qws, __bf16* __restrict__ kws,
    __bf16* __restrict__ vtws) {
  const int tid = threadIdx.x, wave = tid >> 5, lane = tid & 31;
  const int col = lane & 15, hl = lane >> 4, r0 = hl * 8;
  const int tile = blockIdx.x * 8 + wave;
  if (tile >= MT32 * 36) return;
  const int mt = tile / 36, nt = tile % 36;
  const int m0 = mt * 32, n0 = nt * 64;

  v8f acc[2][4];
#pragma unroll
  for (int u = 0; u < 2; ++u)
#pragma unroll
    for (int j = 0; j < 4; ++j) acc[u][j] = zero8();

  const __bf16* ap0 = xbf + (size_t)(m0 + col) * CDIM;
  const __bf16* ap1 = xbf + (size_t)(m0 + 16 + col) * CDIM;

  for (int kk = 0; kk < CDIM; kk += 32) {
    v16bf a0 = load_bf16_tile(ap0 + kk, hl);
    v16bf a1 = load_bf16_tile(ap1 + kk, hl);
#pragma unroll
    for (int j = 0; j < 4; ++j) {
      v16bf bt = load_bf16_tile(wbf + (size_t)(n0 + j * 16 + col) * CDIM + kk, hl);
      acc[0][j] = wmma_bf16(a0, bt, acc[0][j]);
      acc[1][j] = wmma_bf16(a1, bt, acc[1][j]);
    }
  }

#pragma unroll
  for (int u = 0; u < 2; ++u) {
#pragma unroll
    for (int j = 0; j < 4; ++j) {
      const int n = n0 + j * 16 + col;
#pragma unroll
      for (int e = 0; e < 8; ++e) {
        const int m = m0 + u * 16 + r0 + e;
        float v = acc[u][j][e];
        const int bb = m / NTOK, tok = m % NTOK;
        if (n < 768) {                         // Q: bias + scale hd^-0.5
          v = (v + qb[n]) * 0.125f;
          const int head = n >> 6, d = n & 63;
          qws[(((size_t)bb * HEADS + head) * NTOK + tok) * HD + d] = f2bf(v);
        } else if (n < 1536) {                 // K: zero bias
          const int nd = n - 768, head = nd >> 6, d = nd & 63;
          kws[(((size_t)bb * HEADS + head) * NTOK + tok) * HD + d] = f2bf(v);
        } else {                               // V: bias, store transposed
          const int nd = n - 1536, head = nd >> 6, d = nd & 63;
          v += vb[nd];
          vtws[(((size_t)bb * HEADS + head) * HD + d) * VSTR + tok] = f2bf(v);
        }
      }
    }
  }
}

// ---------------------------------------------------------------------------
// Kernel 2: fused flash attention.  Block = 4 waves, one 16-row q-tile per
// wave; 32-key blocks staged in LDS (shared by all 4 waves).
// ---------------------------------------------------------------------------
__global__ __launch_bounds__(128) void st_attn_kernel(
    const __bf16* __restrict__ qws, const __bf16* __restrict__ kws,
    const __bf16* __restrict__ vtws, const float* __restrict__ bsp,
    const float* __restrict__ rtt, __bf16* __restrict__ aows) {
  __shared__ __bf16 Kt[32 * 72];      // [token][hd]  stride 72 (pad)
  __shared__ __bf16 Vt[64 * 72];      // [hd][token]  stride 72
  __shared__ __bf16 Pt[4][16 * 72];   // per-wave P staging

  const int tid = threadIdx.x;
  const int wave = tid >> 5, lane = tid & 31;
  const int col = lane & 15, hl = lane >> 4, r0 = hl * 8;

  const int h = blockIdx.y, b = blockIdx.z;
  const int bh = b * HEADS + h;

  const int qtile_raw = blockIdx.x * 4 + wave;
  const bool store_ok = (qtile_raw < QTILES);
  const int qtile = store_ok ? qtile_raw : (QTILES - 1);
  const int q0 = qtile * 16;

  // Resident Q A-tiles (hd 0..31 and 32..63)
  const __bf16* qbase = qws + (size_t)bh * NTOK * HD;
  int qrow = q0 + col; if (qrow > NTOK - 1) qrow = NTOK - 1;
  const __bf16* qp = qbase + (size_t)qrow * HD;
  const v16bf a0 = load_bf16_tile(qp, hl);
  const v16bf a1 = load_bf16_tile(qp + 32, hl);

  int qmod[8], qf[8];
#pragma unroll
  for (int e = 0; e < 8; ++e) {
    int qa = q0 + r0 + e; if (qa > NTOK - 1) qa = NTOK - 1;
    qmod[e] = qa % AREA1; qf[e] = qa / AREA1;
  }

  float m_r[8], l_r[8];
  v8f o[4];
#pragma unroll
  for (int e = 0; e < 8; ++e) { m_r[e] = -3.0e38f; l_r[e] = 0.f; }
#pragma unroll
  for (int j = 0; j < 4; ++j) o[j] = zero8();

  const __bf16* kbase = kws + (size_t)bh * NTOK * HD;
  const __bf16* vbase = vtws + (size_t)bh * HD * VSTR;
  const float* bsp_h = bsp + (size_t)h * AREA1 * AREA1;

  for (int kb = 0; kb < KITERS; ++kb) {
    const int k0 = kb * 32;
    if (kb) __syncthreads();                 // previous tile fully consumed

    { // cooperative staging: K tile (32 tok x 64 hd), V^T tile (64 hd x 32 tok)
      int tok = tid >> 2, seg = tid & 3;
      int gt = k0 + tok; if (gt > NTOK - 1) gt = NTOK - 1;   // masked later
      const uint4* ks = (const uint4*)(kbase + (size_t)gt * HD + seg * 16);
      uint4 d0 = ks[0], d1 = ks[1];
      uint4* kd = (uint4*)(&Kt[tok * 72 + seg * 16]);
      kd[0] = d0; kd[1] = d1;

      int jv = tid >> 1, s2 = tid & 1;
      const uint4* vs = (const uint4*)(vbase + (size_t)jv * VSTR + k0 + s2 * 16);
      uint4 e0 = vs[0], e1 = vs[1];
      uint4* vd = (uint4*)(&Vt[jv * 72 + s2 * 16]);
      vd[0] = e0; vd[1] = e1;
    }
    __syncthreads();

    // S = Q K^T  (two 16x16 key tiles)
    v8f s[2];
#pragma unroll
    for (int n = 0; n < 2; ++n) {
      const __bf16* kp = &Kt[(n * 16 + col) * 72];
      v16bf b0 = load_bf16_tile(kp, hl);
      v16bf b1 = load_bf16_tile(kp + 32, hl);
      v8f accn = wmma_bf16(a0, b0, zero8());
      s[n] = wmma_bf16(a1, b1, accn);
    }

    // + relative-position / temporal bias, tail mask
#pragma unroll
    for (int n = 0; n < 2; ++n) {
      const int kabs = k0 + n * 16 + col;
      if (kabs < NTOK) {
        const int kmod = kabs % AREA1, kf = kabs / AREA1;
        const float* brow = bsp_h + kmod;
#pragma unroll
        for (int e = 0; e < 8; ++e)
          s[n][e] += brow[qmod[e] * AREA1] + rtt[(qf[e] - kf + 7) * HEADS + h];
      } else {
#pragma unroll
        for (int e = 0; e < 8; ++e) s[n][e] = -3.0e38f;
      }
    }

    // online softmax (row stats across the 16 column lanes)
    float ps[2][8];
#pragma unroll
    for (int e = 0; e < 8; ++e) {
      float rm = fmaxf(s[0][e], s[1][e]);
      rm = fmaxf(rm, __shfl_xor(rm, 1, 16));
      rm = fmaxf(rm, __shfl_xor(rm, 2, 16));
      rm = fmaxf(rm, __shfl_xor(rm, 4, 16));
      rm = fmaxf(rm, __shfl_xor(rm, 8, 16));
      const float mn = fmaxf(m_r[e], rm);
      const float alpha = __expf(m_r[e] - mn);
      m_r[e] = mn;
      const float p0 = __expf(s[0][e] - mn);
      const float p1 = __expf(s[1][e] - mn);
      float rs = p0 + p1;
      rs += __shfl_xor(rs, 1, 16);
      rs += __shfl_xor(rs, 2, 16);
      rs += __shfl_xor(rs, 4, 16);
      rs += __shfl_xor(rs, 8, 16);
      l_r[e] = l_r[e] * alpha + rs;
#pragma unroll
      for (int j = 0; j < 4; ++j) o[j][e] *= alpha;
      ps[0][e] = p0; ps[1][e] = p1;
    }

    // P: C-layout -> LDS -> A-layout (wave-private, DS is in-order per wave)
    __bf16* pw = Pt[wave];
#pragma unroll
    for (int n = 0; n < 2; ++n)
#pragma unroll
      for (int e = 0; e < 8; ++e)
        pw[(r0 + e) * 72 + n * 16 + col] = f2bf(ps[n][e]);
    asm volatile("s_wait_dscnt 0" ::: "memory");
    const v16bf pa = load_bf16_tile(&pw[col * 72], hl);

    // O += P V
#pragma unroll
    for (int j = 0; j < 4; ++j) {
      const __bf16* vp = &Vt[(j * 16 + col) * 72];
      v16bf bv = load_bf16_tile(vp, hl);
      o[j] = wmma_bf16(pa, bv, o[j]);
    }
  }

  if (store_ok) {
#pragma unroll
    for (int e = 0; e < 8; ++e) {
      const int qa = q0 + r0 + e;
      if (qa < NTOK) {
        const float inv = (l_r[e] > 0.f) ? 1.0f / l_r[e] : 0.f;   // nan_to_num
        const size_t rowoff = (((size_t)b * NTOK + qa) * HEADS + h) * HD;
#pragma unroll
        for (int j = 0; j < 4; ++j)
          aows[rowoff + j * 16 + col] = f2bf(o[j][e] * inv);
      }
    }
  }
}

// ---------------------------------------------------------------------------
// Kernel 3: output projection GEMM (NT, all-bf16 operands), per-wave 32x64,
// fp32 out + bias.
// ---------------------------------------------------------------------------
__global__ __launch_bounds__(256) void st_proj_kernel(
    const __bf16* __restrict__ aow, const __bf16* __restrict__ wbf,
    const float* __restrict__ pb, float* __restrict__ out) {
  const int tid = threadIdx.x, wave = tid >> 5, lane = tid & 31;
  const int col = lane & 15, hl = lane >> 4, r0 = hl * 8;
  const int tile = blockIdx.x * 8 + wave;
  if (tile >= MT32 * 12) return;
  const int mt = tile / 12, nt = tile % 12;
  const int m0 = mt * 32, n0 = nt * 64;

  v8f acc[2][4];
#pragma unroll
  for (int u = 0; u < 2; ++u)
#pragma unroll
    for (int j = 0; j < 4; ++j) acc[u][j] = zero8();

  const __bf16* ap0 = aow + (size_t)(m0 + col) * CDIM;
  const __bf16* ap1 = aow + (size_t)(m0 + 16 + col) * CDIM;

  for (int kk = 0; kk < CDIM; kk += 32) {
    v16bf a0 = load_bf16_tile(ap0 + kk, hl);
    v16bf a1 = load_bf16_tile(ap1 + kk, hl);
#pragma unroll
    for (int j = 0; j < 4; ++j) {
      v16bf bt = load_bf16_tile(wbf + (size_t)(n0 + j * 16 + col) * CDIM + kk, hl);
      acc[0][j] = wmma_bf16(a0, bt, acc[0][j]);
      acc[1][j] = wmma_bf16(a1, bt, acc[1][j]);
    }
  }

#pragma unroll
  for (int u = 0; u < 2; ++u) {
#pragma unroll
    for (int j = 0; j < 4; ++j) {
      const int n = n0 + j * 16 + col;
      const float bias = pb[n];
#pragma unroll
      for (int e = 0; e < 8; ++e) {
        const int m = m0 + u * 16 + r0 + e;
        out[(size_t)m * CDIM + n] = acc[u][j][e] + bias;
      }
    }
  }
}

// ---------------------------------------------------------------------------
extern "C" void kernel_launch(void* const* d_in, const int* in_sizes, int n_in,
                              void* d_out, int out_size, void* d_ws, size_t ws_size,
                              hipStream_t stream) {
  (void)in_sizes; (void)n_in; (void)out_size; (void)ws_size;
  const float* x      = (const float*)d_in[0];
  const float* qkv_w  = (const float*)d_in[1];
  const float* q_bias = (const float*)d_in[2];
  const float* v_bias = (const float*)d_in[3];
  const float* rpt    = (const float*)d_in[4];
  const float* rtt    = (const float*)d_in[5];
  const float* proj_w = (const float*)d_in[6];
  const float* proj_b = (const float*)d_in[7];
  // d_in[8] = num_frames (static == 8 here)

  char* ws = (char*)d_ws;
  size_t off = 0;
  auto alloc = [&](size_t bytes) {
    void* p = ws + off;
    off += (bytes + 255) & ~(size_t)255;
    return p;
  };
  __bf16* qws  = (__bf16*)alloc((size_t)BATCH * HEADS * NTOK * HD * 2);
  __bf16* kws  = (__bf16*)alloc((size_t)BATCH * HEADS * NTOK * HD * 2);
  __bf16* vtws = (__bf16*)alloc((size_t)BATCH * HEADS * HD * VSTR * 2);
  __bf16* aow  = (__bf16*)alloc((size_t)BATCH * NTOK * CDIM * 2);
  float*  bsp  = (float*)alloc((size_t)HEADS * AREA1 * AREA1 * 4);
  __bf16* xbf  = (__bf16*)alloc((size_t)MROWS * CDIM * 2);
  __bf16* wbf  = (__bf16*)alloc((size_t)3 * CDIM * CDIM * 2);
  __bf16* pwbf = (__bf16*)alloc((size_t)CDIM * CDIM * 2);

  const int n8_x  = MROWS * CDIM / 8;       // 605184
  const int n8_w  = 3 * CDIM * CDIM / 8;    // 221184
  const int n8_pw = CDIM * CDIM / 8;        // 73728

  st_cvt_kernel<<<(n8_x + 255) / 256, 256, 0, stream>>>(x, xbf, n8_x);
  st_cvt_kernel<<<(n8_w + 255) / 256, 256, 0, stream>>>(qkv_w, wbf, n8_w);
  st_cvt_kernel<<<(n8_pw + 255) / 256, 256, 0, stream>>>(proj_w, pwbf, n8_pw);
  st_bias_kernel<<<(AREA1 * AREA1 + 255) / 256, 256, 0, stream>>>(rpt, bsp);

  st_qkv_kernel<<<(MT32 * 36 + 7) / 8, 256, 0, stream>>>(xbf, wbf, q_bias, v_bias,
                                                         qws, kws, vtws);
  st_attn_kernel<<<dim3((QTILES + 3) / 4, HEADS, BATCH), 128, 0, stream>>>(
      qws, kws, vtws, bsp, rtt, aow);
  st_proj_kernel<<<(MT32 * 12 + 7) / 8, 256, 0, stream>>>(aow, pwbf, proj_b,
                                                          (float*)d_out);
}
